// Attention_84275848282258
// MI455X (gfx1250) — matrix operand order repaired and checked by reading, not compiled, not run on hardware
//
#include <hip/hip_runtime.h>

// ---------------------------------------------------------------------------
// Types for CDNA5 WMMA (wave32): v16bf A/B fragments, v8f C/D accumulators.
// ---------------------------------------------------------------------------
typedef __bf16 bf16_t;
typedef bf16_t v16bf __attribute__((ext_vector_type(16)));
typedef float  v8f   __attribute__((ext_vector_type(8)));
typedef unsigned short ushort_t;

__device__ __forceinline__ float bf2f(ushort_t u) {
    union { float f; unsigned int i; } c; c.i = ((unsigned int)u) << 16; return c.f;
}
__device__ __forceinline__ ushort_t f2bf(float f) {
    union { float f; unsigned int i; } c; c.f = f;
    unsigned int i = c.i;
    i += 0x7fffu + ((i >> 16) & 1u);        // round-to-nearest-even
    return (ushort_t)(i >> 16);
}

__device__ __forceinline__ v8f wmma_bf16(v16bf a, v16bf b, v8f c) {
    return __builtin_amdgcn_wmma_f32_16x16x32_bf16(
        /*neg_a=*/false, a, /*neg_b=*/false, b,
        /*c_mod=*/(short)0, c, /*reuse_a=*/false, /*reuse_b=*/false);
}

// A-matrix 16x32 bf16 fragment (row-major [m][k], k contiguous).
// Lanes 0-15: M=lane, K halves {0..7, 16..23}; lanes 16-31: K {8..15, 24..31}.
__device__ __forceinline__ v16bf load_afrag(const ushort_t* base, int stride, int lane) {
    int m = lane & 15;
    int o = (lane & 16) ? 8 : 0;
    const ushort_t* p = base + m * stride + o;
    union { v16bf v; uint4 q[2]; } u;
    u.q[0] = *(const uint4*)(p);
    u.q[1] = *(const uint4*)(p + 16);
    return u.v;
}

// B-matrix 32x16 bf16 fragment from [n][k] layout (k contiguous per n-row).
// Lanes 0-15: N=lane, K {0..7, 16..23}; lanes 16-31: N=lane-16, K {8..15, 24..31}.
__device__ __forceinline__ v16bf load_bfrag(const ushort_t* base, int stride, int lane) {
    int n = lane & 15;
    int o = (lane & 16) ? 8 : 0;
    const ushort_t* p = base + n * stride + o;
    union { v16bf v; uint4 q[2]; } u;
    u.q[0] = *(const uint4*)(p);
    u.q[1] = *(const uint4*)(p + 16);
    return u.v;
}

// Async global->LDS 16-byte copy, tracked by ASYNCcnt (CDNA5 path).
__device__ __forceinline__ void async_copy_b128(const void* gptr, void* lptr) {
    unsigned int lds = (unsigned int)(unsigned long long)lptr;  // low 32 = LDS offset
    unsigned long long ga = (unsigned long long)gptr;
    asm volatile("global_load_async_to_lds_b128 %0, %1, off"
                 :: "v"(lds), "v"(ga) : "memory");
}
__device__ __forceinline__ void wait_asynccnt0() {
    asm volatile("s_wait_asynccnt 0x0" ::: "memory");
}

// ---------------------------------------------------------------------------
// Generic tiled GEMM: Out[M,Ncol] = A[M,Kd] * W[Kd,Ncol]
//   A: fp32 or bf16(row-major), W: fp32 row-major (converted to bf16 in LDS,
//   transposed to [n][k]), Out: bf16 or fp32.
// Block tile 128x128, BK=32, 8 waves (4x2) each 32x64 (2x4 WMMA subtiles).
// ---------------------------------------------------------------------------
template <bool A_IS_F32, bool OUT_IS_F32>
__global__ __launch_bounds__(256) void gemm_wmma(const void* __restrict__ Av,
                                                 const float* __restrict__ W,
                                                 void* __restrict__ Outv,
                                                 int M, int Kd, int Ncol) {
    constexpr int AST = 40;   // LDS A row stride (elements), 80B, 16B-aligned
    constexpr int BST = 40;   // LDS B row stride
    __shared__ __align__(16) ushort_t As[128 * AST];
    __shared__ __align__(16) ushort_t Bs[128 * BST];

    const int tid  = threadIdx.x;
    const int wid  = tid >> 5;
    const int lane = tid & 31;
    const int m0   = blockIdx.y * 128;
    const int n0   = blockIdx.x * 128;
    const int waveM = wid >> 1;   // 0..3 -> 32 rows each
    const int waveN = wid & 1;    // 0..1 -> 64 cols each

    const v8f vzero = {0.f, 0.f, 0.f, 0.f, 0.f, 0.f, 0.f, 0.f};
    v8f acc[2][4];
#pragma unroll
    for (int i = 0; i < 2; ++i)
#pragma unroll
        for (int j = 0; j < 4; ++j) acc[i][j] = vzero;

    const int ksteps = Kd >> 5;
    for (int kt = 0; kt < ksteps; ++kt) {
        __syncthreads();
        // Stage A tile 128x32 -> bf16 LDS [m][k]
        {
            const int row = tid >> 1;
            const int kh  = (tid & 1) << 4;
            const long grow = (long)(m0 + row);
            if (A_IS_F32) {
                const float* A = (const float*)Av;
                const float* src = A + grow * Kd + kt * 32 + kh;
#pragma unroll
                for (int i = 0; i < 16; ++i) As[row * AST + kh + i] = f2bf(src[i]);
            } else {
                const ushort_t* A = (const ushort_t*)Av;
                const ushort_t* src = A + grow * Kd + kt * 32 + kh;
#pragma unroll
                for (int i = 0; i < 16; ++i) As[row * AST + kh + i] = src[i];
            }
        }
        // Stage B tile 32x128 fp32 -> bf16 LDS transposed [n][k]
        {
#pragma unroll
            for (int e = 0; e < 16; ++e) {
                int lin = tid * 16 + e;       // 0..4095
                int n   = lin & 127;
                int kk  = lin >> 7;           // 0..31
                float val = W[(long)(kt * 32 + kk) * Ncol + n0 + n];
                Bs[n * BST + kk] = f2bf(val);
            }
        }
        __syncthreads();

        v16bf afr[2], bfr[4];
#pragma unroll
        for (int i = 0; i < 2; ++i)
            afr[i] = load_afrag(As + (32 * waveM + 16 * i) * AST, AST, lane);
#pragma unroll
        for (int j = 0; j < 4; ++j)
            bfr[j] = load_bfrag(Bs + (64 * waveN + 16 * j) * BST, BST, lane);
#pragma unroll
        for (int i = 0; i < 2; ++i)
#pragma unroll
            for (int j = 0; j < 4; ++j)
                acc[i][j] = wmma_bf16(afr[i], bfr[j], acc[i][j]);
    }

    // Epilogue: C layout — VGPR r holds row (r + 8*(lane>>4)), col (lane&15).
    const int hi = lane >> 4;
    const int nn = lane & 15;
#pragma unroll
    for (int i = 0; i < 2; ++i) {
#pragma unroll
        for (int j = 0; j < 4; ++j) {
#pragma unroll
            for (int r = 0; r < 8; ++r) {
                int row = m0 + 32 * waveM + 16 * i + r + 8 * hi;
                int col = n0 + 64 * waveN + 16 * j + nn;
                float val = acc[i][j][r];
                if (OUT_IS_F32)
                    ((float*)Outv)[(long)row * Ncol + col] = val;
                else
                    ((ushort_t*)Outv)[(long)row * Ncol + col] = f2bf(val);
            }
        }
    }
}

// ---------------------------------------------------------------------------
// RoPE (in-place on bf16, pairs h and h+64 of H=128).
// x: [B, L, heads, 128]; one thread per (b,l,head,h<64).
// ---------------------------------------------------------------------------
__global__ void rope_kernel(ushort_t* __restrict__ x, const int* __restrict__ pos,
                            int Bv, int L, int heads) {
    long tid = (long)blockIdx.x * blockDim.x + threadIdx.x;
    long total = (long)Bv * L * heads * 64;
    if (tid >= total) return;
    int h = (int)(tid & 63);
    long rest = tid >> 6;
    int head = (int)(rest % heads); rest /= heads;
    int l = (int)(rest % L);
    int b = (int)(rest / L);

    float p  = (float)pos[(long)b * L + l];
    float ts = __powf(10000.f, (float)h * (1.0f / 64.0f));
    float ang = p / ts;
    float s, c;
    __sincosf(ang, &s, &c);

    long base = (((long)b * L + l) * heads + head) * 128 + h;
    float x1 = bf2f(x[base]);
    float x2 = bf2f(x[base + 64]);
    x[base]      = f2bf(x1 * c - x2 * s);
    x[base + 64] = f2bf(x2 * c + x1 * s);
}

// ---------------------------------------------------------------------------
// Flash attention (GQA): per block = (b, query head n, 128 query rows).
// q: [B,T,N,H] bf16 (roped), k/v: [B,S,K,H] bf16, attn out: [B,T,N,H] bf16.
// T=S=2048, N=16, K=4, H=128, scale=1.0, no mask. S-block = 64.
// K block staged with GLOBAL_LOAD_ASYNC_TO_LDS_B128 (ASYNCcnt).
// ---------------------------------------------------------------------------
__global__ __launch_bounds__(256) void flash_attn(const ushort_t* __restrict__ q,
                                                  const ushort_t* __restrict__ k,
                                                  const ushort_t* __restrict__ v,
                                                  ushort_t* __restrict__ attn) {
    constexpr int T = 2048, S = 2048, NQ = 16, KH = 4, H = 128;
    constexpr int SB  = 64;   // S-block
    constexpr int KST = 136;  // K block LDS stride (272B, 16B-aligned)
    constexpr int VST = 72;   // Vt LDS stride (144B)
    constexpr int PST = 72;   // P LDS stride
    __shared__ __align__(16) ushort_t Kb[SB * KST];      // [s(64)][h(128)]
    __shared__ __align__(16) ushort_t Vt[128 * VST];     // [h(128)][s(64)]
    __shared__ __align__(16) ushort_t Pb[8 * 16 * PST];  // per-wave 16x64 P

    const int tid  = threadIdx.x;
    const int wid  = tid >> 5;
    const int lane = tid & 31;
    const int hi   = lane >> 4;
    const int nn   = lane & 15;

    const int qt = blockIdx.x;        // T/128 tiles
    const int n  = blockIdx.y;        // query head
    const int b  = blockIdx.z;
    const int kv = n >> 2;            // G = 4
    const int t0 = qt * 128 + wid * 16;

    // Q fragments: 16 rows x 128 H, 4 K-steps of 32.
    const ushort_t* qbase = q + (((long)(b * T + t0) * NQ + n) * H);
    v16bf qa[4];
#pragma unroll
    for (int kq = 0; kq < 4; ++kq)
        qa[kq] = load_afrag(qbase + kq * 32, NQ * H, lane);

    const v8f vzero = {0.f, 0.f, 0.f, 0.f, 0.f, 0.f, 0.f, 0.f};
    v8f O[8];
#pragma unroll
    for (int j = 0; j < 8; ++j) O[j] = vzero;
    float mrun[8], lrun[8];
#pragma unroll
    for (int r = 0; r < 8; ++r) { mrun[r] = -1e30f; lrun[r] = 0.f; }

    for (int s0 = 0; s0 < S; s0 += SB) {
        __syncthreads();   // previous block's Kb/Vt reads complete

        // Async-stage K block [64][128]: 1024 16B chunks, 4 per thread.
#pragma unroll
        for (int i = 0; i < 4; ++i) {
            int c   = tid + 256 * i;
            int row = c >> 4;          // 0..63
            int col = (c & 15) << 3;   // 0,8,..,120 (ushorts)
            const ushort_t* src = k + (((long)(b * S + s0 + row) * KH + kv) * H) + col;
            async_copy_b128(src, Kb + row * KST + col);
        }
        // Stage V block transposed: Vt[h][s]. 32 elems/thread.
        {
            const int s  = tid >> 2;          // 0..63
            const int h0 = (tid & 3) << 5;    // 0,32,64,96
            const ushort_t* src = v + (((long)(b * S + s0 + s) * KH + kv) * H) + h0;
#pragma unroll
            for (int i = 0; i < 32; ++i) Vt[(h0 + i) * VST + s] = src[i];
        }
        wait_asynccnt0();
        __syncthreads();   // Kb (async) + Vt (ds) visible to all waves

        // Scores: 16 x 64 (four 16x16 tiles), K over H=128. 16 WMMAs.
        v8f sc[4];
#pragma unroll
        for (int j = 0; j < 4; ++j) sc[j] = vzero;
#pragma unroll
        for (int j = 0; j < 4; ++j)
#pragma unroll
            for (int kq = 0; kq < 4; ++kq)
                sc[j] = wmma_bf16(qa[kq],
                                  load_bfrag(Kb + (j * 16) * KST + kq * 32, KST, lane),
                                  sc[j]);

        // Online softmax (row spread across the 16 lanes of this half-wave).
        float pr[4][8], scl[8];
#pragma unroll
        for (int r = 0; r < 8; ++r) {
            float cmax = fmaxf(fmaxf(sc[0][r], sc[1][r]), fmaxf(sc[2][r], sc[3][r]));
#pragma unroll
            for (int off = 1; off < 16; off <<= 1)
                cmax = fmaxf(cmax, __shfl_xor(cmax, off, 32));
            float mnew = fmaxf(mrun[r], cmax);
            scl[r] = __expf(mrun[r] - mnew);
            float rs = 0.f;
#pragma unroll
            for (int j = 0; j < 4; ++j) {
                float p = __expf(sc[j][r] - mnew);
                pr[j][r] = p;
                rs += p;
            }
#pragma unroll
            for (int off = 1; off < 16; off <<= 1)
                rs += __shfl_xor(rs, off, 32);
            lrun[r] = lrun[r] * scl[r] + rs;
            mrun[r] = mnew;
        }
#pragma unroll
        for (int j = 0; j < 8; ++j)
#pragma unroll
            for (int r = 0; r < 8; ++r)
                O[j][r] *= scl[r];

        // P -> per-wave LDS tile [row][s], then re-read as A-fragments.
        ushort_t* Pw = Pb + wid * 16 * PST;
#pragma unroll
        for (int j = 0; j < 4; ++j)
#pragma unroll
            for (int r = 0; r < 8; ++r)
                Pw[(r + 8 * hi) * PST + j * 16 + nn] = f2bf(pr[j][r]);
        asm volatile("s_wait_dscnt 0x0" ::: "memory");

        v16bf pa[2];
#pragma unroll
        for (int kk = 0; kk < 2; ++kk)
            pa[kk] = load_afrag(Pw + kk * 32, PST, lane);
        // PV: 8 h-tiles x 2 k-steps = 16 WMMAs.
#pragma unroll
        for (int j = 0; j < 8; ++j)
#pragma unroll
            for (int kk = 0; kk < 2; ++kk)
                O[j] = wmma_bf16(pa[kk],
                                 load_bfrag(Vt + (j * 16) * VST + kk * 32, VST, lane),
                                 O[j]);
    }

    // Normalize and store bf16 [B,T,N,H].
    float inv[8];
#pragma unroll
    for (int r = 0; r < 8; ++r) inv[r] = 1.0f / lrun[r];
#pragma unroll
    for (int j = 0; j < 8; ++j) {
#pragma unroll
        for (int r = 0; r < 8; ++r) {
            int rowt = t0 + r + 8 * hi;
            int h    = j * 16 + nn;
            attn[(((long)(b * T + rowt) * NQ + n) * H) + h] = f2bf(O[j][r] * inv[r]);
        }
    }
}

// ---------------------------------------------------------------------------
// Host-side launch. Inputs (setup_inputs order):
//  0:Xq[f32 2,2048,2048] 1:Xkv[f32 2,2048,2048] 2:q_pos[i32 2,2048]
//  3:kv_pos[i32 2,2048]  4:Wq[f32 2048,16,128]  5:Wk[f32 2048,4,128]
//  6:Wv[f32 2048,4,128]  7:Wo[f32 16,128,2048]
// Output: [2,2048,2048] f32.
// ---------------------------------------------------------------------------
extern "C" void kernel_launch(void* const* d_in, const int* in_sizes, int n_in,
                              void* d_out, int out_size, void* d_ws, size_t ws_size,
                              hipStream_t stream) {
    const float* Xq  = (const float*)d_in[0];
    const float* Xkv = (const float*)d_in[1];
    const int* qpos  = (const int*)d_in[2];
    const int* kpos  = (const int*)d_in[3];
    const float* Wq  = (const float*)d_in[4];
    const float* Wk  = (const float*)d_in[5];
    const float* Wv  = (const float*)d_in[6];
    const float* Wo  = (const float*)d_in[7];
    float* out = (float*)d_out;

    constexpr int Bb = 2, T = 2048, S = 2048, NQ = 16, KH = 4, H = 128;
    constexpr int M = Bb * T;                          // 4096
    constexpr long q_elems = (long)Bb * T * NQ * H;    // 8.39M
    constexpr long k_elems = (long)Bb * S * KH * H;    // 2.10M

    char* ws = (char*)d_ws;
    ushort_t* qbuf    = (ushort_t*)(ws);
    ushort_t* kbuf    = (ushort_t*)(ws + q_elems * 2);
    ushort_t* vbuf    = (ushort_t*)(ws + q_elems * 2 + k_elems * 2);
    ushort_t* attnbuf = (ushort_t*)(ws + q_elems * 2 + 2 * k_elems * 2);

    // Projections (A fp32 -> bf16 out).
    gemm_wmma<true, false><<<dim3(2048 / 128, M / 128), 256, 0, stream>>>(
        Xq, Wq, qbuf, M, 2048, 2048);
    gemm_wmma<true, false><<<dim3(512 / 128, M / 128), 256, 0, stream>>>(
        Xkv, Wk, kbuf, M, 2048, 512);
    gemm_wmma<true, false><<<dim3(512 / 128, M / 128), 256, 0, stream>>>(
        Xkv, Wv, vbuf, M, 2048, 512);

    // RoPE on q (16 heads) and k (4 heads).
    {
        long tq = (long)Bb * T * NQ * 64;
        rope_kernel<<<(unsigned)((tq + 255) / 256), 256, 0, stream>>>(qbuf, qpos, Bb, T, NQ);
        long tk = (long)Bb * S * KH * 64;
        rope_kernel<<<(unsigned)((tk + 255) / 256), 256, 0, stream>>>(kbuf, kpos, Bb, S, KH);
    }

    // Flash attention.
    flash_attn<<<dim3(T / 128, NQ, Bb), 256, 0, stream>>>(qbuf, kbuf, vbuf, attnbuf);

    // Output projection (A bf16 -> fp32 out).
    gemm_wmma<false, true><<<dim3(2048 / 128, M / 128), 256, 0, stream>>>(
        attnbuf, Wo, out, M, 2048, 2048);
}
